// AdaptSelfAttention_43404939493746
// MI455X (gfx1250) — matrix operand order repaired and checked by reading, not compiled, not run on hardware
//
#include <hip/hip_runtime.h>
#include <hip/hip_bf16.h>
#include <math.h>

// ---------------------------------------------------------------------------
// Transformer-XL relative self-attention, fused flash-style for gfx1250.
// B=4, L=1024, H=512, NH=8, HD=64.  Matmuls on v_wmma_f32_16x16x32_f16;
// K/V tiles staged to LDS with global_load_async_to_lds_b128 (ASYNCcnt).
// ---------------------------------------------------------------------------

typedef __attribute__((ext_vector_type(16))) _Float16 v16h;
typedef __attribute__((ext_vector_type(8)))  float    v8f;

#define B_  4
#define L_  1024
#define H_  512
#define NH_ 8
#define HD_ 64
#define RL_ 2048   // rel rows

__device__ __forceinline__ v8f wmma16(v16h a, v16h b, v8f c) {
  // D = A(16x32 f16) * B(32x16 f16) + C(16x16 f32)
  return __builtin_amdgcn_wmma_f32_16x16x32_f16(
      /*neg_a=*/false, a, /*neg_b=*/false, b,
      /*c_mod=*/(short)0, c, /*reuse_a=*/false, /*reuse_b=*/false);
}

// async copy 16B global -> LDS (GV mode), tracked by ASYNCcnt
__device__ __forceinline__ void async_g2l_b128(unsigned int lds_off,
                                               const void* gaddr) {
  asm volatile("global_load_async_to_lds_b128 %0, %1, off"
               :: "v"(lds_off), "v"(gaddr) : "memory");
}
__device__ __forceinline__ void wait_async0() {
  asm volatile("s_wait_asynccnt 0" ::: "memory");
}

// ---- fragment loaders (CDNA5 16-bit VGPR layouts, ISA 7.12.2) --------------
// A 16x32: lane m = lane&15; half=lane>>4; vgpr-pair p: kb = 2*(p&3)+(p>=4?16:0)+8*half
__device__ __forceinline__ v16h ldA(const _Float16* p, int sm, int sk, int lane) {
  v16h a;
  int m = lane & 15, half = lane >> 4;
  const _Float16* row = p + m * sm;
#pragma unroll
  for (int pr = 0; pr < 8; ++pr) {
    int kb = 2 * (pr & 3) + (pr >= 4 ? 16 : 0) + 8 * half;
    a[2 * pr]     = row[kb * sk];
    a[2 * pr + 1] = row[(kb + 1) * sk];
  }
  return a;
}

// B 32x16: lane n = lane&15; vgpr j holds K = kb,kb+1, kb = 2*(j&3)+(j>=4?16:0)+8*half
__device__ __forceinline__ v16h ldB(const _Float16* p, int sk, int sn, int lane) {
  v16h b;
  int n = lane & 15, half = lane >> 4;
#pragma unroll
  for (int j = 0; j < 8; ++j) {
    int k0 = 2 * (j & 3) + (j >= 4 ? 16 : 0) + 8 * half;
    b[2 * j]     = p[k0 * sk + n * sn];
    b[2 * j + 1] = p[(k0 + 1) * sk + n * sn];
  }
  return b;
}

// B operand whose N dim indexes a (clamped) band of rel rows
__device__ __forceinline__ v16h ldBrel(const _Float16* relh, int base, int nt,
                                       int kb, int lane) {
  v16h b;
  int n = (lane & 15) + nt * 16, half = lane >> 4;
  int row = base + n;
  row = row < 0 ? 0 : (row > RL_ - 1 ? RL_ - 1 : row);
  const _Float16* p = relh + row * HD_ + kb;
#pragma unroll
  for (int j = 0; j < 8; ++j) {
    int k0 = 2 * (j & 3) + (j >= 4 ? 16 : 0) + 8 * half;
    b[2 * j]     = p[k0];
    b[2 * j + 1] = p[k0 + 1];
  }
  return b;
}

// ---------------------------------------------------------------------------
// Kernel 1: C(f16) = A(f32, MxK) * W(f32, KxN) + bias ;  one wave per 16x16 tile
// ---------------------------------------------------------------------------
__global__ __launch_bounds__(256) void proj_gemm(const float* __restrict__ A,
                                                 const float* __restrict__ W,
                                                 const float* __restrict__ bias,
                                                 _Float16* __restrict__ out,
                                                 int M, int N, int K) {
  int wave = (blockIdx.x * blockDim.x + threadIdx.x) >> 5;
  int lane = threadIdx.x & 31;
  int ntiles = N >> 4;
  int tm = wave / ntiles, tn = wave - tm * ntiles;
  int m = lane & 15, half = lane >> 4;
  int row = tm * 16 + m;
  int col = tn * 16 + m;  // B/C column for this lane

  v8f acc = {};
  for (int k0 = 0; k0 < K; k0 += 32) {
    v16h a, b;
    const float* ar = A + (size_t)row * K + k0;
#pragma unroll
    for (int pr = 0; pr < 8; ++pr) {
      int kb = 2 * (pr & 3) + (pr >= 4 ? 16 : 0) + 8 * half;
      a[2 * pr]     = (_Float16)ar[kb];
      a[2 * pr + 1] = (_Float16)ar[kb + 1];
    }
#pragma unroll
    for (int j = 0; j < 8; ++j) {
      int kk = k0 + 2 * (j & 3) + (j >= 4 ? 16 : 0) + 8 * half;
      b[2 * j]     = (_Float16)W[(size_t)kk * N + col];
      b[2 * j + 1] = (_Float16)W[(size_t)(kk + 1) * N + col];
    }
    acc = wmma16(a, b, acc);
  }
  float bv = bias[col];
#pragma unroll
  for (int r = 0; r < 8; ++r) {
    int rr = tm * 16 + r + 8 * half;
    out[(size_t)rr * N + col] = (_Float16)(acc[r] + bv);
  }
}

// ---------------------------------------------------------------------------
// Kernel 2: f32 -> f16 convert (for K)
// ---------------------------------------------------------------------------
__global__ void cvt_f16(const float* __restrict__ in, _Float16* __restrict__ out,
                        int n) {
  int i = blockIdx.x * blockDim.x + threadIdx.x;
  if (i < n) out[i] = (_Float16)in[i];
}

// ---------------------------------------------------------------------------
// Kernel 3: rel = sinusoid_embedding(2048x512) @ w_r_w + w_r_b  (f16 out)
//           Db[n,l] = r_w_bias[n,:] . rel[l,:]                  (f32 out)
// ---------------------------------------------------------------------------
__global__ __launch_bounds__(64) void rel_kernel(const float* __restrict__ w_r_w,
                                                 const float* __restrict__ w_r_b,
                                                 const float* __restrict__ r_w_bias,
                                                 _Float16* __restrict__ relh,
                                                 float* __restrict__ Db) {
  __shared__ float emb[H_];
  __shared__ float rrow[HD_];
  int l = blockIdx.x, t = threadIdx.x;
  float pos = (float)(l - L_);
  const float c = __logf(10000.0f) / 255.0f;  // half-1 = 255
#pragma unroll
  for (int jj = 0; jj < 4; ++jj) {
    int j = t * 4 + jj;                       // j in [0,256)
    float f = __expf(-c * (float)j);
    float ang = pos * f;
    emb[j]        = __sinf(ang);
    emb[256 + j]  = __cosf(ang);
  }
  __syncthreads();
  float acc = w_r_b[t];
  for (int k = 0; k < H_; ++k) acc += emb[k] * w_r_w[k * HD_ + t];
  relh[(size_t)l * HD_ + t] = (_Float16)acc;
  rrow[t] = acc;
  __syncthreads();
  if (t < NH_) {
    float s = 0.f;
    for (int d = 0; d < HD_; ++d) s += r_w_bias[t * HD_ + d] * rrow[d];
    Db[t * RL_ + l] = s;
  }
}

// ---------------------------------------------------------------------------
// Kernel 4: fused relative attention, one wave per (b, head, 16-row q tile).
// K/V pair tiles async-staged into LDS; all matmuls WMMA; online softmax.
// ---------------------------------------------------------------------------
#define WPB 2  // waves per block (LDS: ~36 KB/block)
__global__ __launch_bounds__(WPB * 32) void attn_kernel(
    const _Float16* __restrict__ Qh, const _Float16* __restrict__ Kh,
    const _Float16* __restrict__ Vh, const _Float16* __restrict__ relh,
    const float* __restrict__ Db, const float* __restrict__ r_r_bias,
    const int* __restrict__ seq_len_p, float* __restrict__ out) {
  __shared__ float    bq[WPB][16][48];   // q . rel-band   (per wave)
  __shared__ float    ek[WPB][32][48];   // k . rel-band   (per wave)
  __shared__ _Float16 pst[WPB][16][32];  // probs staging C-layout -> A-layout
  __shared__ _Float16 kt[WPB][32][HD_];  // async-staged K pair tile
  __shared__ _Float16 vt[WPB][32][HD_];  // async-staged V pair tile

  int w = threadIdx.x >> 5, lane = threadIdx.x & 31;
  int gw = blockIdx.x * WPB + w;                  // 0 .. 2047
  int itile = gw & 63;                            // L/16
  int head  = (gw >> 6) & (NH_ - 1);
  int b     = gw >> 9;
  int i0 = itile << 4;
  int m = lane & 15, half = lane >> 4;
  int seq_len = *seq_len_p;

  const _Float16* kl = &kt[w][0][0];
  const _Float16* vl = &vt[w][0][0];
  unsigned int ktoff = (unsigned int)(uintptr_t)kl;
  unsigned int vtoff = (unsigned int)(uintptr_t)vl;

  // --- Q fragments (plain, and +r_r_bias for the AC term), d split 0..31/32..63
  v16h qa0, qa1, qb0, qb1;
  {
    const _Float16* qrow = Qh + ((size_t)b * L_ + i0 + m) * H_ + head * HD_;
    const float* rb = r_r_bias + head * HD_;
#pragma unroll
    for (int pr = 0; pr < 8; ++pr) {
      int kb = 2 * (pr & 3) + (pr >= 4 ? 16 : 0) + 8 * half;
      _Float16 x0 = qrow[kb], x1 = qrow[kb + 1];
      qa0[2 * pr] = x0;               qa0[2 * pr + 1] = x1;
      qb0[2 * pr] = (_Float16)((float)x0 + rb[kb]);
      qb0[2 * pr + 1] = (_Float16)((float)x1 + rb[kb + 1]);
      _Float16 y0 = qrow[32 + kb], y1 = qrow[32 + kb + 1];
      qa1[2 * pr] = y0;               qa1[2 * pr + 1] = y1;
      qb1[2 * pr] = (_Float16)((float)y0 + rb[32 + kb]);
      qb1[2 * pr + 1] = (_Float16)((float)y1 + rb[32 + kb + 1]);
    }
  }

  v8f o0 = {}, o1 = {}, o2 = {}, o3 = {};  // 16x64 output accumulator
  float mrow[8], lrow[8];
#pragma unroll
  for (int r = 0; r < 8; ++r) { mrow[r] = -3.0e38f; lrow[r] = 0.f; }

  for (int pair = 0; pair < L_ / 32; ++pair) {
    int j0 = pair * 32;
    const _Float16* Kjb = Kh + ((size_t)b * L_ + j0) * H_ + head * HD_;
    const _Float16* Vjb = Vh + ((size_t)b * L_ + j0) * H_ + head * HD_;
    __builtin_prefetch(Kjb + 32 * H_, 0, 1);  // warm L2 for next pair

    // ---- async-stage K/V pair tiles (32 rows x 128 B) into per-wave LDS
#pragma unroll
    for (int it = 0; it < 8; ++it) {
      int c = lane + 32 * it;                 // 256 16-byte chunks per tile
      int row = c >> 3;
      int bo = (c & 7) * 16;                  // byte offset within row
      int go = row * (H_ * 2) + bo;           // global row stride = 1024 B
      unsigned int lo = (unsigned int)(row * (HD_ * 2) + bo);
      async_g2l_b128(ktoff + lo, (const char*)Kjb + go);
      async_g2l_b128(vtoff + lo, (const char*)Vjb + go);
    }
    wait_async0();
    __syncthreads();

    // ---- AC = (q + r_r_bias) @ k^T  : two 16x16 tiles, K=64 in two chunks
    v8f s0 = {}, s1 = {};
    s0 = wmma16(qb0, ldB(kl,             1, HD_, lane), s0);
    s0 = wmma16(qb1, ldB(kl + 32,        1, HD_, lane), s0);
    s1 = wmma16(qb0, ldB(kl + 16 * HD_,      1, HD_, lane), s1);
    s1 = wmma16(qb1, ldB(kl + 16 * HD_ + 32, 1, HD_, lane), s1);

    // ---- banded GEMM: bq[il][c] = q[i0+il] . rel[bqbase + c],  c in [0,48)
    int bqbase = L_ + j0 - i0 - 15;
#pragma unroll
    for (int t = 0; t < 3; ++t) {
      v8f c = {};
      c = wmma16(qa0, ldBrel(relh, bqbase, t, 0,  lane), c);
      c = wmma16(qa1, ldBrel(relh, bqbase, t, 32, lane), c);
#pragma unroll
      for (int r = 0; r < 8; ++r) bq[w][r + 8 * half][t * 16 + m] = c[r];
    }

    // ---- banded GEMM: ek[jc][c] = k[j0+jc] . rel[ekbase + c]
    int ekbase = L_ + i0 - j0 - 31;
#pragma unroll
    for (int mt = 0; mt < 2; ++mt) {
      v16h ka0 = ldA(kl + mt * 16 * HD_,      HD_, 1, lane);
      v16h ka1 = ldA(kl + mt * 16 * HD_ + 32, HD_, 1, lane);
#pragma unroll
      for (int t = 0; t < 3; ++t) {
        v8f c = {};
        c = wmma16(ka0, ldBrel(relh, ekbase, t, 0,  lane), c);
        c = wmma16(ka1, ldBrel(relh, ekbase, t, 32, lane), c);
#pragma unroll
        for (int r = 0; r < 8; ++r)
          ek[w][mt * 16 + r + 8 * half][t * 16 + m] = c[r];
      }
    }
    __syncthreads();

    // ---- gather shift terms along diagonals + D_ bias + length mask
    const float* Dbh = Db + head * RL_;
#pragma unroll
    for (int r = 0; r < 8; ++r) {
      int il = r + 8 * half;
      int i = i0 + il;
      {
        int jc = m, j = j0 + jc;
        float v = s0[r] + bq[w][il][15 + jc - il] + Dbh[L_ + j - i] +
                  ek[w][jc][31 + il - jc];
        s0[r] = (j >= seq_len) ? -1.0e15f : v;
      }
      {
        int jc = m + 16, j = j0 + jc;
        float v = s1[r] + bq[w][il][15 + jc - il] + Dbh[L_ + j - i] +
                  ek[w][jc][31 + il - jc];
        s1[r] = (j >= seq_len) ? -1.0e15f : v;
      }
    }

    // ---- online softmax (row reductions across the 16 lanes of each half)
#pragma unroll
    for (int r = 0; r < 8; ++r) {
      float smax = fmaxf(s0[r], s1[r]);
#pragma unroll
      for (int off = 1; off < 16; off <<= 1)
        smax = fmaxf(smax, __shfl_xor(smax, off, 32));
      float mnew = fmaxf(mrow[r], smax);
      float alpha = __expf(mrow[r] - mnew);
      float p0 = __expf(s0[r] - mnew), p1 = __expf(s1[r] - mnew);
      float ps = p0 + p1;
#pragma unroll
      for (int off = 1; off < 16; off <<= 1) ps += __shfl_xor(ps, off, 32);
      lrow[r] = lrow[r] * alpha + ps;
      mrow[r] = mnew;
      o0[r] *= alpha; o1[r] *= alpha; o2[r] *= alpha; o3[r] *= alpha;
      int il = r + 8 * half;
      pst[w][il][m]      = (_Float16)p0;   // C-layout -> LDS for A-layout reload
      pst[w][il][m + 16] = (_Float16)p1;
    }
    __syncthreads();

    // ---- O += P(16x32) @ V(32x64) : 4 WMMA over the d dimension (V from LDS)
    v16h pf = ldA(&pst[w][0][0], 32, 1, lane);
    o0 = wmma16(pf, ldB(vl +  0, HD_, 1, lane), o0);
    o1 = wmma16(pf, ldB(vl + 16, HD_, 1, lane), o1);
    o2 = wmma16(pf, ldB(vl + 32, HD_, 1, lane), o2);
    o3 = wmma16(pf, ldB(vl + 48, HD_, 1, lane), o3);
  }

  // ---- normalize and write out[b, i, head*64 + d]
#pragma unroll
  for (int r = 0; r < 8; ++r) {
    float inv = 1.0f / lrow[r];
    int i = i0 + r + 8 * half;
    float* orow = out + ((size_t)b * L_ + i) * H_ + head * HD_;
    orow[m]      = o0[r] * inv;
    orow[16 + m] = o1[r] * inv;
    orow[32 + m] = o2[r] * inv;
    orow[48 + m] = o3[r] * inv;
  }
}

// ---------------------------------------------------------------------------
extern "C" void kernel_launch(void* const* d_in, const int* in_sizes, int n_in,
                              void* d_out, int out_size, void* d_ws, size_t ws_size,
                              hipStream_t stream) {
  const float* query   = (const float*)d_in[0];
  const float* key     = (const float*)d_in[1];
  const float* value   = (const float*)d_in[2];
  const float* w_q_w   = (const float*)d_in[3];
  const float* w_q_b   = (const float*)d_in[4];
  const float* w_v_w   = (const float*)d_in[5];
  const float* w_v_b   = (const float*)d_in[6];
  const float* w_r_w   = (const float*)d_in[7];
  const float* w_r_b   = (const float*)d_in[8];
  const float* r_r_bias = (const float*)d_in[9];
  const float* r_w_bias = (const float*)d_in[10];
  const int*   seq_len  = (const int*)d_in[11];
  float* out = (float*)d_out;

  // workspace layout (f16 Q/V/K, f16 rel, f32 Db)  ~12.3 MB
  const size_t NQ = (size_t)B_ * L_ * H_;        // 2,097,152
  _Float16* Qh   = (_Float16*)d_ws;
  _Float16* Vh   = Qh + NQ;
  _Float16* Kh   = Vh + NQ;
  _Float16* relh = Kh + NQ;
  float*    Db   = (float*)(relh + (size_t)RL_ * HD_);

  // Q / V projections: (4096x512)@(512x512), 8192 waves, 8 waves/block
  proj_gemm<<<1024, 256, 0, stream>>>(query, w_q_w, w_q_b, Qh, B_ * L_, H_, H_);
  proj_gemm<<<1024, 256, 0, stream>>>(value, w_v_w, w_v_b, Vh, B_ * L_, H_, H_);
  cvt_f16<<<(int)(NQ / 256), 256, 0, stream>>>(key, Kh, (int)NQ);
  rel_kernel<<<RL_, 64, 0, stream>>>(w_r_w, w_r_b, r_w_bias, relh, Db);

  // fused attention: 2048 waves = 4*8*64 (b,head,qtile), WPB waves/block
  attn_kernel<<<2048 / WPB, WPB * 32, 0, stream>>>(Qh, Kh, Vh, relh, Db,
                                                   r_r_bias, seq_len, out);
}
